// SingleHeadAttention_15272903705140
// MI455X (gfx1250) — matrix operand order repaired and checked by reading, compiled, not run on hardware
//
#include <hip/hip_runtime.h>
#include <hip/hip_bf16.h>

typedef __attribute__((ext_vector_type(16))) _Float16 v16h;
typedef __attribute__((ext_vector_type(8)))  _Float16 v8h;
typedef __attribute__((ext_vector_type(8)))  float    v8f;

#define HEAD  256
#define NEMB  1024
#define BATCH 8
#define SEQ   2048
#define SVP   264   // padded LDS row stride (halfs): 528B, 16B-aligned

__device__ __forceinline__ v8f wmma_f16(v16h a, v16h b, v8f c) {
  // D = A(16x32 f16) * B(32x16 f16) + C(16x16 f32)
  return __builtin_amdgcn_wmma_f32_16x16x32_f16(
      /*neg_a=*/false, a, /*neg_b=*/false, b,
      /*c_mod=*/(short)0, c, /*reuse_a=*/false, /*reuse_b=*/false);
}

// A-fragment loader from row-major f16: lane (m,hid) holds row m,
// halfs [0..15] = K(hid*16 .. +15).
__device__ __forceinline__ v16h load_a16(const _Float16* p) {
  v8h lo = *(const v8h*)(p);
  v8h hi = *(const v8h*)(p + 8);
  v16h r;
#pragma unroll
  for (int i = 0; i < 8; ++i) { r[i] = lo[i]; r[8 + i] = hi[i]; }
  return r;
}

// gfx1250 async bulk copy: 16B global -> LDS, tracked by ASYNCcnt.
__device__ __forceinline__ void async_g2l_b128(unsigned lds_byte_addr,
                                               const void* gaddr) {
  asm volatile("global_load_async_to_lds_b128 %0, %1, off"
               :: "v"(lds_byte_addr), "v"(gaddr) : "memory");
}
__device__ __forceinline__ void wait_async0() {
  asm volatile("s_wait_asynccnt 0x0" ::: "memory");
}
__device__ __forceinline__ unsigned lds_addr_of(const void* p) {
  // flat shared pointer: low 32 bits are the LDS byte address
  return (unsigned)(unsigned long long)p;
}

// ---------------------------------------------------------------------------
// Phase 1: Q/K/V = x @ W{q,k,v}, fp32 in -> f16 out (f32 WMMA accumulate).
// grid = (BT/16, HEAD/64, 3), block = 128 (4 waves).
// ---------------------------------------------------------------------------
__global__ __launch_bounds__(128)
void qkv_proj_kernel(const float* __restrict__ x,
                     const float* __restrict__ Wq,
                     const float* __restrict__ Wk,
                     const float* __restrict__ Wv,
                     _Float16* __restrict__ Qh,
                     _Float16* __restrict__ Kh,
                     _Float16* __restrict__ Vh) {
  const int lane = threadIdx.x & 31;
  const int wave = threadIdx.x >> 5;
  const int m    = lane & 15;
  const int hid  = lane >> 4;
  const int row0 = blockIdx.x * 16;
  const int n0   = (blockIdx.y * 4 + wave) * 16;

  const float* W;
  _Float16* Out;
  if (blockIdx.z == 0)      { W = Wq; Out = Qh; }
  else if (blockIdx.z == 1) { W = Wk; Out = Kh; }
  else                      { W = Wv; Out = Vh; }

  v8f acc = {};
  for (int k0 = 0; k0 < NEMB; k0 += 32) {
    // A fragment: x[row0+m, k0 + hid*16 + {0..15}], f32 -> f16
    const float* xr = x + (size_t)(row0 + m) * NEMB + k0 + hid * 16;
    v16h a;
#pragma unroll
    for (int i = 0; i < 8; ++i) { a[i]     = (_Float16)xr[i];
                                  a[8 + i] = (_Float16)xr[8 + i]; }
    // B fragment: b[i] = W[k0 + hid*16 + i, n0 + m] (coalesced across lanes)
    const float* wp = W + (size_t)(k0 + hid * 16) * HEAD + n0 + m;
    v16h bf;
#pragma unroll
    for (int i = 0; i < 16; ++i) bf[i] = (_Float16)wp[(size_t)i * HEAD];

    acc = wmma_f16(a, bf, acc);
  }
  // C/D layout: VGPR r, lane (m,hid) -> row r+8*hid, col m
#pragma unroll
  for (int r = 0; r < 8; ++r)
    Out[(size_t)(row0 + r + 8 * hid) * HEAD + n0 + m] = (_Float16)acc[r];
}

// ---------------------------------------------------------------------------
// Phase 2: causal flash attention. grid = (SEQ/16, BATCH), block = 128.
// One block = one 16-row Q tile; KV chunks of 64 cols; wave w computes the
// S tile at cols kv0+w*16 and owns output columns [w*64, w*64+64).
// V chunk is staged to LDS via gfx1250 async B128 copies, overlapped with
// the S-tile WMMAs and the softmax update.
// ---------------------------------------------------------------------------
__global__ __launch_bounds__(128)
void attn_kernel(const _Float16* __restrict__ Qh,
                 const _Float16* __restrict__ Kh,
                 const _Float16* __restrict__ Vh,
                 float* __restrict__ out) {
  __shared__ __align__(16) _Float16 sV[64][SVP];  // V chunk (async-staged)
  __shared__ __align__(16) float    sS[16][64];   // scores chunk (f32)
  __shared__ __align__(16) _Float16 sP[16][64];   // softmax probs (f16)
  __shared__ float sRed[16][8];
  __shared__ float sM[16], sL[16], sAlpha[16];

  const int tid  = threadIdx.x;
  const int lane = tid & 31;
  const int wave = tid >> 5;
  const int m    = lane & 15;
  const int hid  = lane >> 4;
  const int b    = blockIdx.y;
  const int q0   = blockIdx.x * 16;

  const _Float16* Qb = Qh + (size_t)b * SEQ * HEAD;
  const _Float16* Kb = Kh + (size_t)b * SEQ * HEAD;
  const _Float16* Vb = Vh + (size_t)b * SEQ * HEAD;

  // Q fragments for this 16-row tile (8 fragments cover H=256 K-dim)
  v16h qfrag[8];
#pragma unroll
  for (int f = 0; f < 8; ++f)
    qfrag[f] = load_a16(Qb + (size_t)(q0 + m) * HEAD + f * 32 + hid * 16);

  v8f oacc[4] = {};  // this wave's O columns [wave*64, wave*64+64)

  if (tid < 16) { sM[tid] = -__builtin_inff(); sL[tid] = 0.f; }
  __syncthreads();

  const float scale = 0.0625f;  // 1/sqrt(256)
  const int kvEnd = q0 + 16;

  for (int kv0 = 0; kv0 < kvEnd; kv0 += 64) {
    // ---- async-stage V chunk [kv0..kv0+64) x HEAD into LDS.
    // 2048 16B units; thread t copies units {j*128 + t}.
#pragma unroll
    for (int j = 0; j < 16; ++j) {
      const int unit = j * 128 + tid;
      const int row  = unit >> 5;         // 0..63
      const int c8   = (unit & 31) * 8;   // half offset within row
      async_g2l_b128(lds_addr_of(&sV[row][c8]),
                     Vb + (size_t)(kv0 + row) * HEAD + c8);
    }
    if (kv0 + 64 < kvEnd)  // prefetch next chunk of K rows
      __builtin_prefetch(Kb + (size_t)(kv0 + 64 + wave * 16 + m) * HEAD, 0, 1);

    // ---- S tile: this wave covers kv cols [c0, c0+16)
    const int c0 = kv0 + wave * 16;
    v8f s = {};
#pragma unroll
    for (int f = 0; f < 8; ++f) {
      // B = K^T: b[i] = Kb[(c0+m)*HEAD + f*32 + hid*16 + i] -> contiguous 32B
      v16h bf = *(const v16h*)(Kb + (size_t)(c0 + m) * HEAD + f * 32 + hid * 16);
      s = wmma_f16(qfrag[f], bf, s);
    }
    // scale + causal mask -> LDS
#pragma unroll
    for (int r = 0; r < 8; ++r) {
      const int row = r + 8 * hid;       // local q row
      const int col = c0 + m;            // global kv col
      float v = s[r] * scale;
      if (col > q0 + row) v = -__builtin_inff();
      sS[row][wave * 16 + m] = v;
    }
    __syncthreads();

    // ---- online softmax over the 64-wide chunk (8 threads per row)
    {
      const int row  = tid >> 3;
      const int tsub = tid & 7;
      float pm = -__builtin_inff();
#pragma unroll
      for (int j = 0; j < 8; ++j) pm = fmaxf(pm, sS[row][tsub * 8 + j]);
      sRed[row][tsub] = pm;
      __syncthreads();
      if (tsub == 0) {
        float cm = sRed[row][0];
#pragma unroll
        for (int j = 1; j < 8; ++j) cm = fmaxf(cm, sRed[row][j]);
        const float mold  = sM[row];
        const float mnew  = fmaxf(mold, cm);
        const float alpha = __expf(mold - mnew);  // 0 on first chunk
        sM[row] = mnew; sAlpha[row] = alpha; sL[row] *= alpha;
      }
      __syncthreads();
      const float mnew = sM[row];
      float psum = 0.f;
#pragma unroll
      for (int j = 0; j < 8; ++j) {
        const int c = tsub * 8 + j;
        const float v = sS[row][c];
        const float e = (v == -__builtin_inff()) ? 0.f : __expf(v - mnew);
        sP[row][c] = (_Float16)e;
        psum += e;
      }
      sRed[row][tsub] = psum;
      __syncthreads();
      if (tsub == 0) {
        float cs = 0.f;
#pragma unroll
        for (int j = 0; j < 8; ++j) cs += sRed[row][j];
        sL[row] += cs;
      }
    }

    // ---- wait for this wave's async V copies, then sync all waves
    wait_async0();
    __syncthreads();

    // ---- rescale O accumulators by alpha, then O += P @ V (V from LDS)
#pragma unroll
    for (int r = 0; r < 8; ++r) {
      const float alpha = sAlpha[r + 8 * hid];
#pragma unroll
      for (int t = 0; t < 4; ++t) oacc[t][r] *= alpha;
    }
#pragma unroll
    for (int kc = 0; kc < 2; ++kc) {
      v16h pa = load_a16(&sP[m][kc * 32 + hid * 16]);  // P fragment from LDS
#pragma unroll
      for (int t = 0; t < 4; ++t) {
        const int n0 = wave * 64 + t * 16;
        // V B-frag from LDS: b[i] = sV[kc*32 + hid*16 + i][n0 + m]
        const _Float16* vp = &sV[kc * 32 + hid * 16][n0 + m];
        v16h vb;
#pragma unroll
        for (int i = 0; i < 16; ++i) vb[i] = vp[(size_t)i * SVP];
        oacc[t] = wmma_f16(pa, vb, oacc[t]);
      }
    }
    __syncthreads();  // protect sS/sP/sAlpha/sV before next chunk
  }

  // ---- normalize by running sum and store fp32 output
  float* ob = out + (size_t)b * SEQ * HEAD;
#pragma unroll
  for (int r = 0; r < 8; ++r) {
    const int row = r + 8 * hid;
    const float inv = 1.0f / sL[row];
#pragma unroll
    for (int t = 0; t < 4; ++t)
      ob[(size_t)(q0 + row) * HEAD + wave * 64 + t * 16 + m] = oacc[t][r] * inv;
  }
}

// ---------------------------------------------------------------------------
extern "C" void kernel_launch(void* const* d_in, const int* in_sizes, int n_in,
                              void* d_out, int out_size, void* d_ws, size_t ws_size,
                              hipStream_t stream) {
  const float* x  = (const float*)d_in[0];
  const float* Wq = (const float*)d_in[1];
  const float* Wk = (const float*)d_in[2];
  const float* Wv = (const float*)d_in[3];
  float* out = (float*)d_out;

  const size_t elems = (size_t)BATCH * SEQ * HEAD;  // 4M f16 = 8MB each
  _Float16* Qh = (_Float16*)d_ws;
  _Float16* Kh = Qh + elems;
  _Float16* Vh = Kh + elems;

  dim3 g1(BATCH * SEQ / 16, HEAD / 64, 3);  // (1024, 4, 3)
  qkv_proj_kernel<<<g1, 128, 0, stream>>>(x, Wq, Wk, Wv, Qh, Kh, Vh);

  dim3 g2(SEQ / 16, BATCH);                 // (128, 8)
  attn_kernel<<<g2, 128, 0, stream>>>(Qh, Kh, Vh, out);
}
